// GPT_42339787604742
// MI455X (gfx1250) — matrix operand order repaired and checked by reading, not compiled, not run on hardware
//
#include <hip/hip_runtime.h>
#include <hip/hip_bf16.h>
#include <math.h>

#define L_    4
#define D_    512
#define H_    8
#define DH_   64
#define FF_   2048
#define V_    32000
#define B_    2
#define T_    1024
#define WD_   32
#define BAND_ 128
#define SCALE_ 0.125f   /* 1/sqrt(64) */

typedef __bf16 bf16_t;
typedef __attribute__((ext_vector_type(16))) __bf16 v16bf;
typedef __attribute__((ext_vector_type(8)))  float  v8f;
typedef __attribute__((ext_vector_type(4)))  int    v4i;

// CDNA5 async global->LDS copies (ASYNCcnt) if the toolchain declares them.
#if defined(__HIP_DEVICE_COMPILE__) && defined(__gfx1250__) && \
    __has_builtin(__builtin_amdgcn_global_load_async_to_lds_b128) && \
    __has_builtin(__builtin_amdgcn_s_wait_asynccnt)
#define HAVE_ASYNC_LDS 1
#else
#define HAVE_ASYNC_LDS 0
#endif

__device__ __forceinline__ void lds_copy16(const bf16_t* g, bf16_t* l)
{
#if HAVE_ASYNC_LDS
    __builtin_amdgcn_global_load_async_to_lds_b128(
        (__attribute__((address_space(1))) v4i*)g,
        (__attribute__((address_space(3))) v4i*)l, 0, 0);
#else
    *(float4*)l = *(const float4*)g;
#endif
}

// ---------------------------------------------------------------------------
// Batched GEMM:  C[M,N] = alpha * A[M,K] * B^T + bias,  B stored [N,K] (weights)
// A,B bf16; C fp32. 64x64 tile per 128-thread (4-wave32) block.
// Double-buffered LDS staging via async global->LDS copies.
// Batch z -> (b=z/nh, h=z%nh) with independent pointer strides.
// Requires M%64==0, N%64==0, K%32==0, row strides multiples of 8 elements.
// ---------------------------------------------------------------------------
__global__ __launch_bounds__(128)
void gemm_bf16_kernel(const bf16_t* __restrict__ A, const bf16_t* __restrict__ Bm,
                      float* __restrict__ C, const float* __restrict__ bias,
                      int K, int lda, int ldb, int ldc, float alpha,
                      long sAb, long sAh, long sBb, long sBh, long sCb, long sCh,
                      int nh)
{
    __shared__ alignas(16) bf16_t ldsA[2][64 * 40];
    __shared__ alignas(16) bf16_t ldsB[2][64 * 40];

    const int z  = blockIdx.z;
    const int zb = z / nh;
    const int zh = z - zb * nh;
    A  += (size_t)zb * sAb + (size_t)zh * sAh;
    Bm += (size_t)zb * sBb + (size_t)zh * sBh;
    C  += (size_t)zb * sCb + (size_t)zh * sCh;

    const int m0   = blockIdx.y * 64;
    const int n0   = blockIdx.x * 64;
    const int tid  = threadIdx.x;
    const int lane = tid & 31;          // wave32 lane
    const int wv   = tid >> 5;          // wave id -> 16-row strip of the tile
    const int half = lane >> 4;
    const int lm   = lane & 15;

    // staging coordinates (hoisted out of the K loop): each thread copies
    // 16B for rows (sr, sr+32) of both tiles.
    const int sr = tid >> 2;            // 0..31
    const int sc = (tid & 3) << 3;      // 0,8,16,24
    const bf16_t* pa0 = A  + (size_t)(m0 + sr) * lda + sc;
    const bf16_t* pa1 = pa0 + (size_t)32 * lda;
    const bf16_t* pb0 = Bm + (size_t)(n0 + sr) * ldb + sc;
    const bf16_t* pb1 = pb0 + (size_t)32 * ldb;
    const int lo0 = sr * 40 + sc;
    const int lo1 = (sr + 32) * 40 + sc;

    v8f acc[4];
    acc[0] = (v8f)(0.0f); acc[1] = (v8f)(0.0f);
    acc[2] = (v8f)(0.0f); acc[3] = (v8f)(0.0f);

    auto issue = [&](int k0, int buf) {
        lds_copy16(pa0 + k0, &ldsA[buf][lo0]);
        lds_copy16(pa1 + k0, &ldsA[buf][lo1]);
        lds_copy16(pb0 + k0, &ldsB[buf][lo0]);
        lds_copy16(pb1 + k0, &ldsB[buf][lo1]);
    };

    const int nk = K >> 5;
    issue(0, 0);

    for (int kt = 0; kt < nk; ++kt) {
        const int cur = kt & 1;
        if (kt + 1 < nk) {
            issue((kt + 1) << 5, cur ^ 1);   // prefetch next tile (other buffer)
#if HAVE_ASYNC_LDS
            __builtin_amdgcn_s_wait_asynccnt(4);   // tile kt complete (4 in flight ok)
#endif
        } else {
#if HAVE_ASYNC_LDS
            __builtin_amdgcn_s_wait_asynccnt(0);
#endif
        }
        __syncthreads();

        const bf16_t* lA = ldsA[cur];
        const bf16_t* lB = ldsB[cur];

        // A fragment (ISA 16-bit A 16x32 layout): two contiguous 16B runs
        v16bf a;
        const int am = wv * 16 + lm;
#pragma unroll
        for (int v = 0; v < 8; ++v) {
            int kk = ((v & 4) << 2) + half * 8 + ((v & 3) << 1);
            a[2 * v]     = lA[am * 40 + kk];
            a[2 * v + 1] = lA[am * 40 + kk + 1];
        }
#pragma unroll
        for (int nt = 0; nt < 4; ++nt) {
            v16bf bv;
            const int bn = nt * 16 + lm;
#pragma unroll
            for (int v = 0; v < 8; ++v) {
                int kk = ((v & 4) << 2) + half * 8 + ((v & 3) << 1);
                bv[2 * v]     = lB[bn * 40 + kk];
                bv[2 * v + 1] = lB[bn * 40 + kk + 1];
            }
            acc[nt] = __builtin_amdgcn_wmma_f32_16x16x32_bf16(
                false, a, false, bv, (short)0, acc[nt], false, false);
        }
        __syncthreads();   // protect buffers before next issue overwrites
    }

    // epilogue: C layout lane -> N=lm, VGPR r -> M = r + 8*half
#pragma unroll
    for (int nt = 0; nt < 4; ++nt) {
        int cn = n0 + nt * 16 + lm;
        float bb = bias ? bias[cn] : 0.0f;
#pragma unroll
        for (int r = 0; r < 8; ++r) {
            int cm = m0 + wv * 16 + r + (half << 3);
            C[(size_t)cm * ldc + cn] = acc[nt][r] * alpha + bb;
        }
    }
}

// ---------------------------------------------------------------------------
// LayerNorm over D=512, one 128-thread block per token row, bf16 output.
// ---------------------------------------------------------------------------
__global__ __launch_bounds__(128)
void ln_bf16_kernel(const float* __restrict__ x, const float* __restrict__ g,
                    const float* __restrict__ b, bf16_t* __restrict__ out)
{
    __shared__ float red[128];
    const int row = blockIdx.x, tid = threadIdx.x;
    const float* xr = x + (size_t)row * D_;

    float s = 0.0f;
    for (int i = tid; i < D_; i += 128) s += xr[i];
    red[tid] = s; __syncthreads();
    for (int o = 64; o; o >>= 1) { if (tid < o) red[tid] += red[tid + o]; __syncthreads(); }
    float mean = red[0] / (float)D_;
    __syncthreads();

    float v = 0.0f;
    for (int i = tid; i < D_; i += 128) { float d = xr[i] - mean; v += d * d; }
    red[tid] = v; __syncthreads();
    for (int o = 64; o; o >>= 1) { if (tid < o) red[tid] += red[tid + o]; __syncthreads(); }
    float rstd = rsqrtf(red[0] / (float)D_ + 1e-5f);

    bf16_t* orow = out + (size_t)row * D_;
    for (int i = tid; i < D_; i += 128)
        orow[i] = (bf16_t)((xr[i] - mean) * rstd * g[i] + b[i]);
}

// ---------------------------------------------------------------------------
// DAPE pointwise MLP over logits + ALiBi + causal/banded mask, in place.
// One thread per (b,i,j); cat = [att, zeros] so only first H columns matter.
// ---------------------------------------------------------------------------
__global__ __launch_bounds__(256)
void dape_kernel(float* __restrict__ att,
                 const float* __restrict__ gw, const float* __restrict__ gb,
                 const float* __restrict__ vw, const float* __restrict__ vb,
                 const float* __restrict__ ow, const float* __restrict__ ob)
{
    __shared__ float sgw[WD_ * 16], svw[WD_ * 16], sow[H_ * WD_];
    __shared__ float sgb[WD_], svb[WD_], sob[H_];
    const int tid = threadIdx.x;
    for (int t = tid; t < WD_ * 16; t += 256) { sgw[t] = gw[t]; svw[t] = vw[t]; }
    for (int t = tid; t < H_ * WD_;  t += 256) sow[t] = ow[t];
    if (tid < WD_) { sgb[tid] = gb[tid]; svb[tid] = vb[tid]; }
    if (tid < H_)  sob[tid] = ob[tid];
    __syncthreads();

    long idx = (long)blockIdx.x * 256 + tid;   // < B*T*T exactly
    int  j = (int)(idx % T_);
    long r = idx / T_;
    int  i = (int)(r % T_);
    int  b = (int)(r / T_);
    size_t base = (size_t)b * H_ * T_ * T_ + (size_t)i * T_ + j;

    float av[H_], lb[H_];
#pragma unroll
    for (int h = 0; h < H_; ++h) av[h] = att[base + (size_t)h * T_ * T_];
#pragma unroll
    for (int h = 0; h < H_; ++h) lb[h] = sob[h];

#pragma unroll 4
    for (int w = 0; w < WD_; ++w) {
        float sg = sgb[w], sv = svb[w];
#pragma unroll
        for (int h = 0; h < H_; ++h) {
            sg += av[h] * sgw[w * 16 + h];
            sv += av[h] * svw[w * 16 + h];
        }
        float hid = (sg / (1.0f + __expf(-sg))) * sv;   // silu(sg)*sv
#pragma unroll
        for (int h = 0; h < H_; ++h) lb[h] += hid * sow[h * WD_ + w];
    }

    const int d = i - j;
#pragma unroll
    for (int h = 0; h < H_; ++h) {
        float slope = exp2f(-(float)(h + 1));           // 0.5^(h+1)
        bool ok = (d >= 0) && (((h & 1) == 0) || (d < BAND_));
        float val = av[h] + lb[h] - slope * (float)d;
        att[base + (size_t)h * T_ * T_] = ok ? val : -__builtin_inff();
    }
}

// ---------------------------------------------------------------------------
// Row softmax over T=1024, one wave32 per row, bf16 probs out.
// ---------------------------------------------------------------------------
__global__ __launch_bounds__(256)
void softmax_kernel(const float* __restrict__ att, bf16_t* __restrict__ outp)
{
    const int lane = threadIdx.x & 31;
    const int row  = blockIdx.x * 8 + (threadIdx.x >> 5);   // B*H*T rows
    const float* src = att + (size_t)row * T_;

    float ev[T_ / 32];
    float mx = -__builtin_inff();
#pragma unroll
    for (int c = 0; c < T_ / 32; ++c) {
        float v = src[lane + c * 32];
        ev[c] = v; mx = fmaxf(mx, v);
    }
#pragma unroll
    for (int o = 16; o; o >>= 1) mx = fmaxf(mx, __shfl_xor(mx, o, 32));

    float sum = 0.0f;
#pragma unroll
    for (int c = 0; c < T_ / 32; ++c) { float e = __expf(ev[c] - mx); ev[c] = e; sum += e; }
#pragma unroll
    for (int o = 16; o; o >>= 1) sum += __shfl_xor(sum, o, 32);

    float rinv = 1.0f / sum;
    bf16_t* dst = outp + (size_t)row * T_;
#pragma unroll
    for (int c = 0; c < T_ / 32; ++c) dst[lane + c * 32] = (bf16_t)(ev[c] * rinv);
}

// ---------------------------------------------------------------------------
// Elementwise helpers
// ---------------------------------------------------------------------------
__global__ void cvt_kernel(const float* __restrict__ s, bf16_t* __restrict__ d, long n)
{
    long i = (long)blockIdx.x * 256 + threadIdx.x;
    if (i < n) d[i] = (bf16_t)s[i];
}

__global__ void add_kernel(float* __restrict__ x, const float* __restrict__ c, long n)
{
    long i = (long)blockIdx.x * 256 + threadIdx.x;
    if (i < n) x[i] += c[i];
}

__global__ void silu_mul_kernel(const float* __restrict__ g, const float* __restrict__ v,
                                bf16_t* __restrict__ o, long n)
{
    long i = (long)blockIdx.x * 256 + threadIdx.x;
    if (i < n) {
        float gg = g[i];
        o[i] = (bf16_t)((gg / (1.0f + __expf(-gg))) * v[i]);
    }
}

__global__ void embed_kernel(const int* __restrict__ idxp, const float* __restrict__ wte,
                             const float* __restrict__ wpe, float* __restrict__ x)
{
    long i = (long)blockIdx.x * 256 + threadIdx.x;   // < B*T*D exactly
    int  d  = (int)(i % D_);
    long bt = i / D_;
    int  t  = (int)(bt % T_);
    x[i] = wte[(size_t)idxp[bt] * D_ + d] + wpe[(size_t)t * D_ + d];
}

// vt[b,h,d,t] = (bf16) v[b,t,h,d]   (v = qkv[..., 2D + h*DH + d])
__global__ void vtrans_kernel(const float* __restrict__ qkv, bf16_t* __restrict__ vt)
{
    long i = (long)blockIdx.x * 256 + threadIdx.x;   // < B*H*DH*T exactly
    int  t = (int)(i % T_);
    long r = i / T_;
    int  d = (int)(r % DH_);
    long r2 = r / DH_;
    int  h = (int)(r2 % H_);
    int  b = (int)(r2 / H_);
    vt[i] = (bf16_t)qkv[((size_t)b * T_ + t) * (3 * D_) + 2 * D_ + h * DH_ + d];
}

// ---------------------------------------------------------------------------
// Host side
// ---------------------------------------------------------------------------
static inline void launch_gemm(const bf16_t* A, const bf16_t* Bm, float* C,
                               const float* bias, int M, int N, int K,
                               int lda, int ldb, int ldc, float alpha,
                               long sAb, long sAh, long sBb, long sBh,
                               long sCb, long sCh, int nh, int nbatch,
                               hipStream_t s)
{
    dim3 grid(N / 64, M / 64, nbatch), blk(128);
    gemm_bf16_kernel<<<grid, blk, 0, s>>>(A, Bm, C, bias, K, lda, ldb, ldc,
                                          alpha, sAb, sAh, sBb, sBh, sCb, sCh, nh);
}

extern "C" void kernel_launch(void* const* d_in, const int* in_sizes, int n_in,
                              void* d_out, int out_size, void* d_ws, size_t ws_size,
                              hipStream_t stream)
{
    (void)in_sizes; (void)n_in; (void)out_size; (void)ws_size;

    const int*   idx     = (const int*)  d_in[0];
    const float* wte     = (const float*)d_in[1];
    const float* wpe     = (const float*)d_in[2];
    const float* ln1_g   = (const float*)d_in[3];
    const float* ln1_b   = (const float*)d_in[4];
    const float* attn_w  = (const float*)d_in[5];
    const float* attn_b  = (const float*)d_in[6];
    const float* proj_w  = (const float*)d_in[7];
    const float* proj_b  = (const float*)d_in[8];
    const float* dape_gw = (const float*)d_in[9];
    const float* dape_gb = (const float*)d_in[10];
    const float* dape_vw = (const float*)d_in[11];
    const float* dape_vb = (const float*)d_in[12];
    const float* dape_ow = (const float*)d_in[13];
    const float* dape_ob = (const float*)d_in[14];
    const float* ln2_g   = (const float*)d_in[15];
    const float* ln2_b   = (const float*)d_in[16];
    const float* gate_w  = (const float*)d_in[17];
    const float* gate_b  = (const float*)d_in[18];
    const float* val_w   = (const float*)d_in[19];
    const float* val_b   = (const float*)d_in[20];
    const float* mlp_w   = (const float*)d_in[21];
    const float* mlp_b   = (const float*)d_in[22];
    const float* lnf_g   = (const float*)d_in[23];
    const float* lnf_b   = (const float*)d_in[24];
    const float* lm_w    = (const float*)d_in[25];
    float* out = (float*)d_out;

    // ---- workspace layout ----
    char* ws = (char*)d_ws;
    size_t off = 0;
    auto alloc = [&](size_t bytes) -> char* {
        char* p = ws + off;
        off += (bytes + 255) & ~(size_t)255;
        return p;
    };
    float*  x    = (float*) alloc((size_t)B_ * T_ * D_ * 4);
    bf16_t* hbf  = (bf16_t*)alloc((size_t)B_ * T_ * D_ * 2);
    float*  qkv  = (float*) alloc((size_t)B_ * T_ * 3 * D_ * 4);
    bf16_t* qkvb = (bf16_t*)alloc((size_t)B_ * T_ * 3 * D_ * 2);
    bf16_t* vt   = (bf16_t*)alloc((size_t)B_ * H_ * DH_ * T_ * 2);
    float*  att  = (float*) alloc((size_t)B_ * H_ * T_ * T_ * 4);
    bf16_t* attp = (bf16_t*)alloc((size_t)B_ * H_ * T_ * T_ * 2);
    float*  ybuf = (float*) alloc((size_t)B_ * T_ * D_ * 4);
    bf16_t* ybf  = (bf16_t*)alloc((size_t)B_ * T_ * D_ * 2);
    float*  gbuf = (float*) alloc((size_t)B_ * T_ * FF_ * 4);   // also proj/mlp C
    float*  vbuf = (float*) alloc((size_t)B_ * T_ * FF_ * 4);
    bf16_t* ffbf = (bf16_t*)alloc((size_t)B_ * T_ * FF_ * 2);
    bf16_t* wbf  = (bf16_t*)alloc((size_t)V_ * D_ * 2);         // weight staging (max = lm_w)

    const long nXD = (long)B_ * T_ * D_;

    embed_kernel<<<nXD / 256, 256, 0, stream>>>(idx, wte, wpe, x);

    for (int l = 0; l < L_; ++l) {
        // ---- attention ----
        ln_bf16_kernel<<<B_ * T_, 128, 0, stream>>>(x, ln1_g + l * D_, ln1_b + l * D_, hbf);

        long nw = (long)3 * D_ * D_;
        cvt_kernel<<<(nw + 255) / 256, 256, 0, stream>>>(attn_w + (size_t)l * nw, wbf, nw);
        launch_gemm(hbf, wbf, qkv, attn_b + (size_t)l * 3 * D_,
                    B_ * T_, 3 * D_, D_, D_, D_, 3 * D_, 1.0f,
                    0, 0, 0, 0, 0, 0, 1, 1, stream);

        long nq = (long)B_ * T_ * 3 * D_;
        cvt_kernel<<<(nq + 255) / 256, 256, 0, stream>>>(qkv, qkvb, nq);
        vtrans_kernel<<<nXD / 256, 256, 0, stream>>>(qkv, vt);

        // att = scale * q @ k^T, batched over (b,h)
        launch_gemm(qkvb, qkvb + D_, att, nullptr,
                    T_, T_, DH_, 3 * D_, 3 * D_, T_, SCALE_,
                    (long)T_ * 3 * D_, DH_, (long)T_ * 3 * D_, DH_,
                    (long)H_ * T_ * T_, (long)T_ * T_, H_, B_ * H_, stream);

        dape_kernel<<<(int)((long)B_ * T_ * T_ / 256), 256, 0, stream>>>(
            att,
            dape_gw + (size_t)l * WD_ * 2 * H_, dape_gb + (size_t)l * WD_,
            dape_vw + (size_t)l * WD_ * 2 * H_, dape_vb + (size_t)l * WD_,
            dape_ow + (size_t)l * H_ * WD_,     dape_ob + (size_t)l * H_);

        softmax_kernel<<<B_ * H_ * T_ / 8, 256, 0, stream>>>(att, attp);

        // y = p @ v : B^T form with vt[b,h,d,t], write into (B,T,H*DH)
        launch_gemm(attp, vt, ybuf, nullptr,
                    T_, DH_, T_, T_, T_, D_, 1.0f,
                    (long)H_ * T_ * T_, (long)T_ * T_,
                    (long)H_ * DH_ * T_, (long)DH_ * T_,
                    (long)T_ * D_, DH_, H_, B_ * H_, stream);

        cvt_kernel<<<(nXD + 255) / 256, 256, 0, stream>>>(ybuf, ybf, nXD);
        long np = (long)D_ * D_;
        cvt_kernel<<<(np + 255) / 256, 256, 0, stream>>>(proj_w + (size_t)l * np, wbf, np);
        launch_gemm(ybf, wbf, gbuf, proj_b + (size_t)l * D_,
                    B_ * T_, D_, D_, D_, D_, D_, 1.0f,
                    0, 0, 0, 0, 0, 0, 1, 1, stream);
        add_kernel<<<nXD / 256, 256, 0, stream>>>(x, gbuf, nXD);

        // ---- MLP ----
        ln_bf16_kernel<<<B_ * T_, 128, 0, stream>>>(x, ln2_g + l * D_, ln2_b + l * D_, hbf);

        long nf = (long)FF_ * D_;
        cvt_kernel<<<(nf + 255) / 256, 256, 0, stream>>>(gate_w + (size_t)l * nf, wbf, nf);
        launch_gemm(hbf, wbf, gbuf, gate_b + (size_t)l * FF_,
                    B_ * T_, FF_, D_, D_, D_, FF_, 1.0f,
                    0, 0, 0, 0, 0, 0, 1, 1, stream);
        cvt_kernel<<<(nf + 255) / 256, 256, 0, stream>>>(val_w + (size_t)l * nf, wbf, nf);
        launch_gemm(hbf, wbf, vbuf, val_b + (size_t)l * FF_,
                    B_ * T_, FF_, D_, D_, D_, FF_, 1.0f,
                    0, 0, 0, 0, 0, 0, 1, 1, stream);

        long nFF = (long)B_ * T_ * FF_;
        silu_mul_kernel<<<nFF / 256, 256, 0, stream>>>(gbuf, vbuf, ffbf, nFF);

        cvt_kernel<<<(nf + 255) / 256, 256, 0, stream>>>(mlp_w + (size_t)l * nf, wbf, nf);
        launch_gemm(ffbf, wbf, gbuf, mlp_b + (size_t)l * D_,
                    B_ * T_, D_, FF_, FF_, FF_, D_, 1.0f,
                    0, 0, 0, 0, 0, 0, 1, 1, stream);
        add_kernel<<<nXD / 256, 256, 0, stream>>>(x, gbuf, nXD);
    }

    // ---- final LN + LM head ----
    ln_bf16_kernel<<<B_ * T_, 128, 0, stream>>>(x, lnf_g, lnf_b, hbf);
    long nl = (long)V_ * D_;
    cvt_kernel<<<(nl + 255) / 256, 256, 0, stream>>>(lm_w, wbf, nl);
    launch_gemm(hbf, wbf, out, nullptr,
                B_ * T_, V_, D_, D_, D_, V_, 1.0f,
                0, 0, 0, 0, 0, 0, 1, 1, stream);
}
